// MambaLayer_88252987998581
// MI455X (gfx1250) — matrix operand order repaired and checked by reading, not compiled, not run on hardware
//
#include <hip/hip_runtime.h>
#include <cstdint>
#include <cstddef>

typedef __attribute__((ext_vector_type(16))) __bf16 v16bf;
typedef __attribute__((ext_vector_type(8)))  float  v8f;

#define NB   8
#define CC   96
#define LL   16384
#define DIN  192
#define DST  16
#define NCHUNK 64
#define CHSZ   256

union AFrag { v16bf v; uint32_t u[8]; };
union BFrag { v16bf v; uint4 q[2]; };

__device__ __forceinline__ float bf2f(__bf16 x) { return (float)x; }
__device__ __forceinline__ __bf16 f2bf(float x) { return (__bf16)x; }
__device__ __forceinline__ float silu(float x) { return x / (1.f + __expf(-x)); }

// Async global->LDS copy (CDNA5, ASYNCcnt-tracked); plain copy on host pass.
#if defined(__gfx1250__)
#define ASYNC_LDS_CP32(ldsp, gp)                                               \
  asm volatile("global_load_async_to_lds_b32 %0, %1, off"                      \
               :: "v"((uint32_t)(uintptr_t)(ldsp)),                            \
                  "v"((uint64_t)(uintptr_t)(gp)) : "memory")
#define ASYNC_WAIT() asm volatile("s_wait_asynccnt 0x0" ::: "memory")
#else
#define ASYNC_LDS_CP32(ldsp, gp) (*(uint32_t*)(ldsp) = *(const uint32_t*)(gp))
#define ASYNC_WAIT()
#endif

// ---------------------------------------------------------------------------
// Weight prep: transpose to K-major bf16 (zero-padded), A = -exp(A_log)
// ---------------------------------------------------------------------------
__global__ void prep_weights(const float* __restrict__ in_w,
                             const float* __restrict__ xproj_w,
                             const float* __restrict__ dtproj_w,
                             const float* __restrict__ out_w,
                             const float* __restrict__ A_log,
                             __bf16* __restrict__ in_wt,    // 96 x 384
                             __bf16* __restrict__ xprojt,   // 192 x 48 (38 valid)
                             __bf16* __restrict__ dtprojt,  // 32 x 192 (6 valid K)
                             __bf16* __restrict__ out_wt,   // 192 x 96
                             float*  __restrict__ Aneg) {   // 192 x 16
  int i = blockIdx.x * 256 + threadIdx.x;
  if (i < 96 * 384) { int k = i / 384, n = i % 384; in_wt[i] = f2bf(in_w[n * 96 + k]); return; }
  i -= 96 * 384;
  if (i < 192 * 48) { int k = i / 48, n = i % 48; xprojt[i] = f2bf(n < 38 ? xproj_w[n * 192 + k] : 0.f); return; }
  i -= 192 * 48;
  if (i < 32 * 192) { int k = i / 192, n = i % 192; dtprojt[i] = f2bf(k < 6 ? dtproj_w[n * 6 + k] : 0.f); return; }
  i -= 32 * 192;
  if (i < 192 * 96) { int k = i / 96, n = i % 96; out_wt[i] = f2bf(out_w[n * 192 + k]); return; }
  i -= 192 * 96;
  if (i < 192 * 16) { Aneg[i] = -__expf(A_log[i]); }
}

// ---------------------------------------------------------------------------
// LayerNorm over C=96 per pixel -> bf16 (M x 96); LDS transpose for coalesced
// stores (48 KB per 256-thread block)
// ---------------------------------------------------------------------------
__global__ void ln_kernel(const float* __restrict__ x, const float* __restrict__ g,
                          const float* __restrict__ b, __bf16* __restrict__ xs) {
  __shared__ __bf16 tile[256 * CC];
  size_t row = (size_t)blockIdx.x * 256 + threadIdx.x;   // n*L + l
  int n = (int)(row / LL), l = (int)(row % LL);
  const float* xr = x + ((size_t)n * CC) * LL + l;
  float s = 0.f, ss = 0.f;
  for (int c = 0; c < CC; ++c) { float v = xr[(size_t)c * LL]; s += v; ss += v * v; }
  float mu = s * (1.f / CC);
  float inv = rsqrtf(ss * (1.f / CC) - mu * mu + 1e-5f);
  __bf16* o = tile + threadIdx.x * CC;
  for (int c = 0; c < CC; ++c)
    o[c] = f2bf((xr[(size_t)c * LL] - mu) * inv * g[c] + b[c]);
  __syncthreads();
  const uint32_t* src = (const uint32_t*)tile;               // 12288 dwords
  uint32_t* dst = (uint32_t*)(xs + (size_t)blockIdx.x * 256 * CC);
  for (int i = threadIdx.x; i < 256 * CC / 2; i += 256) dst[i] = src[i];
}

// ---------------------------------------------------------------------------
// bf16 WMMA GEMM: D = A(MxK) * B(KxN). Each wave computes a 16 x (16*NT)
// strip, reusing one A fragment across NT WMMAs (A-bandwidth / NT).
// MODE 0: fp32 store                 MODE 1: softplus(d+bias[col]) fp32
// MODE 2: transposed (n,Cout,Lseq)   MODE 3: bf16 store
// ---------------------------------------------------------------------------
template <int NT, int MODE>
__global__ void gemm_bf16(const __bf16* __restrict__ A, const __bf16* __restrict__ B,
                          float* __restrict__ D, __bf16* __restrict__ Dbf,
                          int M, int N, int K, const float* __restrict__ bias,
                          int Lseq, int Cout) {
  int wave = blockIdx.x * (blockDim.x >> 5) + (threadIdx.x >> 5);
  int lane = threadIdx.x & 31;
  int stripsN = N / (16 * NT);
  int tot = (M >> 4) * stripsN;
  if (wave >= tot) return;
  int tm = (wave / stripsN) << 4;
  int tn = (wave % stripsN) * (16 * NT);
  int m = lane & 15, hi = lane >> 4;

  v8f acc[NT] = {};
  for (int kk = 0; kk < K; kk += 32) {
    AFrag a;
    const __bf16* Ar = A + (size_t)(tm + m) * K + kk + hi * 8;
    a.u[0] = *(const uint32_t*)(Ar + 0);
    a.u[1] = *(const uint32_t*)(Ar + 2);
    a.u[2] = *(const uint32_t*)(Ar + 4);
    a.u[3] = *(const uint32_t*)(Ar + 6);
    a.u[4] = *(const uint32_t*)(Ar + 16);
    a.u[5] = *(const uint32_t*)(Ar + 18);
    a.u[6] = *(const uint32_t*)(Ar + 20);
    a.u[7] = *(const uint32_t*)(Ar + 22);
    if (kk + 32 < K) __builtin_prefetch(Ar + 32, 0, 0);     // global_prefetch
    const __bf16* Brow = B + (size_t)(kk + hi * 8) * N + tn + m;
    #pragma unroll
    for (int j = 0; j < NT; ++j) {
      BFrag b;
      #pragma unroll
      for (int s = 0; s < 8; ++s) {
        b.v[s] = Brow[(size_t)s * N + j * 16];
        b.v[8 + s] = Brow[(size_t)(16 + s) * N + j * 16];
      }
      acc[j] = __builtin_amdgcn_wmma_f32_16x16x32_bf16(false, a.v, false, b.v,
                                                       (short)0, acc[j], false, false);
    }
  }

  #pragma unroll
  for (int j = 0; j < NT; ++j) {
    int col = tn + j * 16 + m;
    #pragma unroll
    for (int r = 0; r < 8; ++r) {
      int row = tm + r + hi * 8;
      float v = acc[j][r];
      if (MODE == 0) {
        D[(size_t)row * N + col] = v;
      } else if (MODE == 1) {
        v += bias[col];
        D[(size_t)row * N + col] = (v > 20.f) ? v : log1pf(__expf(v));
      } else if (MODE == 2) {
        int nIdx = row / Lseq, l = row % Lseq;
        D[((size_t)nIdx * Cout + col) * Lseq + l] = v;
      } else {
        Dbf[(size_t)row * N + col] = f2bf(v);
      }
    }
  }
}

// ---------------------------------------------------------------------------
// Depthwise causal conv (k=4) + SiLU, reading xin = xz[:, :192] (bf16)
// ---------------------------------------------------------------------------
__global__ void conv_silu(const __bf16* __restrict__ xz, const float* __restrict__ w,
                          const float* __restrict__ cb, __bf16* __restrict__ xc) {
  size_t idx = (size_t)blockIdx.x * 256 + threadIdx.x;   // n*L*192
  int d = (int)(idx % DIN);
  size_t nl = idx / DIN;
  int l = (int)(nl % LL);
  size_t nbase = (nl - l) * 384;                          // (n*L)*384
  float acc = cb[d];
  #pragma unroll
  for (int j = 0; j < 4; ++j) {
    int ls = l - 3 + j;
    if (ls >= 0) acc += w[d * 4 + j] * bf2f(xz[nbase + (size_t)ls * 384 + d]);
  }
  xc[nl * DIN + d] = f2bf(silu(acc));
}

// ---------------------------------------------------------------------------
// Pad dt input: (M x 6) fp32 -> (M x 32) bf16
// ---------------------------------------------------------------------------
__global__ void dt_prep(const float* __restrict__ xdbl, __bf16* __restrict__ dtin) {
  size_t mrow = (size_t)blockIdx.x * 256 + threadIdx.x;
  const float* r = xdbl + mrow * 48;
  __bf16* o = dtin + mrow * 32;
  #pragma unroll
  for (int k = 0; k < 6; ++k)  o[k] = f2bf(r[k]);
  #pragma unroll
  for (int k = 6; k < 32; ++k) o[k] = f2bf(0.f);
}

// ---------------------------------------------------------------------------
// Scan pass 1: per chunk of 256 steps compute (P = prod dA, S = local scan)
// block = (n, chunk); thread = d; 16 states in registers; B async-staged
// ---------------------------------------------------------------------------
__global__ void scan_pass1(const float* __restrict__ dtb, const __bf16* __restrict__ xcb,
                           const float* __restrict__ xdbl, const float* __restrict__ Aneg,
                           float* __restrict__ Pc, float* __restrict__ Sc) {
  int n = blockIdx.x >> 6, chunk = blockIdx.x & 63, d = threadIdx.x;
  __shared__ float Bs[32][17];
  float A_[DST], P[DST], S[DST];
  #pragma unroll
  for (int s = 0; s < DST; ++s) { A_[s] = Aneg[d * DST + s]; P[s] = 1.f; S[s] = 0.f; }
  int l0 = chunk * CHSZ;
  for (int sub = 0; sub < CHSZ / 32; ++sub) {
    int lb = l0 + sub * 32;
    for (int i = threadIdx.x; i < 512; i += DIN) {
      int st = i >> 4, s = i & 15;
      ASYNC_LDS_CP32(&Bs[st][s], &xdbl[((size_t)n * LL + lb + st) * 48 + 6 + s]);
    }
    ASYNC_WAIT();
    __syncthreads();
    for (int t = 0; t < 32; ++t) {
      size_t rix = ((size_t)n * LL + lb + t) * DIN + d;
      float dt = dtb[rix];
      float xv = bf2f(xcb[rix]);
      float dx = dt * xv;
      #pragma unroll
      for (int s = 0; s < DST; ++s) {
        float dA = __expf(dt * A_[s]);
        P[s] *= dA;
        S[s] = dA * S[s] + dx * Bs[t][s];
      }
    }
    __syncthreads();
  }
  size_t base = ((size_t)blockIdx.x * DIN + d) * DST;
  #pragma unroll
  for (int s = 0; s < DST; ++s) { Pc[base + s] = P[s]; Sc[base + s] = S[s]; }
}

// ---------------------------------------------------------------------------
// Scan pass 2: 64-step serial sweep per (n,d,s) -> chunk initial states
// ---------------------------------------------------------------------------
__global__ void scan_pass2(const float* __restrict__ Pc, const float* __restrict__ Sc,
                           float* __restrict__ Hinit) {
  int idx = blockIdx.x * 256 + threadIdx.x;          // (n*192 + d)*16 + s
  int n = idx / (DIN * DST), rem = idx % (DIN * DST);
  float h = 0.f;
  for (int c = 0; c < NCHUNK; ++c) {
    size_t b = ((size_t)(n * NCHUNK + c)) * DIN * DST + rem;
    Hinit[b] = h;
    h = Pc[b] * h + Sc[b];
  }
}

// ---------------------------------------------------------------------------
// Scan pass 3: replay with correct h0, y = sum_s h*C, fuse D-skip + SiLU(z)
// ---------------------------------------------------------------------------
__global__ void scan_pass3(const float* __restrict__ dtb, const __bf16* __restrict__ xcb,
                           const float* __restrict__ xdbl, const float* __restrict__ Aneg,
                           const float* __restrict__ Hinit, const __bf16* __restrict__ xzb,
                           const float* __restrict__ Dsk, __bf16* __restrict__ ybf) {
  int n = blockIdx.x >> 6, chunk = blockIdx.x & 63, d = threadIdx.x;
  __shared__ float Bs[32][17];
  __shared__ float Cs[32][17];
  float A_[DST], h[DST];
  size_t base = ((size_t)blockIdx.x * DIN + d) * DST;
  #pragma unroll
  for (int s = 0; s < DST; ++s) { A_[s] = Aneg[d * DST + s]; h[s] = Hinit[base + s]; }
  float dsk = Dsk[d];
  int l0 = chunk * CHSZ;
  for (int sub = 0; sub < CHSZ / 32; ++sub) {
    int lb = l0 + sub * 32;
    for (int i = threadIdx.x; i < 512; i += DIN) {
      int st = i >> 4, s = i & 15;
      size_t r = ((size_t)n * LL + lb + st) * 48;
      ASYNC_LDS_CP32(&Bs[st][s], &xdbl[r + 6 + s]);
      ASYNC_LDS_CP32(&Cs[st][s], &xdbl[r + 22 + s]);
    }
    ASYNC_WAIT();
    __syncthreads();
    for (int t = 0; t < 32; ++t) {
      size_t rix = ((size_t)n * LL + lb + t) * DIN + d;
      float dt = dtb[rix];
      float xv = bf2f(xcb[rix]);
      float dx = dt * xv;
      float y = 0.f;
      #pragma unroll
      for (int s = 0; s < DST; ++s) {
        float dA = __expf(dt * A_[s]);
        h[s] = dA * h[s] + dx * Bs[t][s];
        y += h[s] * Cs[t][s];
      }
      y += xv * dsk;
      float z = bf2f(xzb[((size_t)n * LL + lb + t) * 384 + DIN + d]);
      ybf[rix] = f2bf(y * silu(z));
    }
    __syncthreads();
  }
}

// ---------------------------------------------------------------------------
extern "C" void kernel_launch(void* const* d_in, const int* in_sizes, int n_in,
                              void* d_out, int out_size, void* d_ws, size_t ws_size,
                              hipStream_t stream) {
  const float* x        = (const float*)d_in[0];
  const float* ln_g     = (const float*)d_in[1];
  const float* ln_b     = (const float*)d_in[2];
  const float* in_w     = (const float*)d_in[3];
  const float* conv_w   = (const float*)d_in[4];
  const float* conv_b   = (const float*)d_in[5];
  const float* xproj_w  = (const float*)d_in[6];
  const float* dtproj_w = (const float*)d_in[7];
  const float* dtproj_b = (const float*)d_in[8];
  const float* A_log    = (const float*)d_in[9];
  const float* D_skip   = (const float*)d_in[10];
  const float* out_w    = (const float*)d_in[11];
  float* out = (float*)d_out;

  const size_t M = (size_t)NB * LL;   // 131072
  char* w = (char*)d_ws;
  size_t off = 0;
  auto alloc = [&](size_t bytes) -> void* {
    void* p = w + off;
    off = (off + bytes + 255) & ~(size_t)255;
    return p;
  };
  __bf16* xs     = (__bf16*)alloc(M * CC * 2);         // LN output
  __bf16* xzb    = (__bf16*)alloc(M * 384 * 2);        // in-proj output
  __bf16* xcb    = (__bf16*)alloc(M * DIN * 2);        // conv+silu output
  float*  xdbl   = (float*) alloc(M * 48 * 4);         // x-proj output (padded)
  __bf16* dtin   = (__bf16*)alloc(M * 32 * 2);         // padded dt GEMM input
  float*  dtb    = (float*) alloc(M * DIN * 4);        // softplus(dt)
  __bf16* ybf    = (__bf16*)alloc(M * DIN * 2);        // gated scan output
  float*  Pc     = (float*) alloc((size_t)NB * NCHUNK * DIN * DST * 4);
  float*  Sc     = (float*) alloc((size_t)NB * NCHUNK * DIN * DST * 4);
  float*  Hinit  = (float*) alloc((size_t)NB * NCHUNK * DIN * DST * 4);
  __bf16* in_wt  = (__bf16*)alloc(96 * 384 * 2);
  __bf16* xprojt = (__bf16*)alloc(192 * 48 * 2);
  __bf16* dtprjt = (__bf16*)alloc(32 * 192 * 2);
  __bf16* out_wt = (__bf16*)alloc(192 * 96 * 2);
  float*  Aneg   = (float*) alloc(192 * 16 * 4);
  (void)ws_size; (void)in_sizes; (void)n_in; (void)out_size;

  prep_weights<<<288, 256, 0, stream>>>(in_w, xproj_w, dtproj_w, out_w, A_log,
                                        in_wt, xprojt, dtprjt, out_wt, Aneg);
  ln_kernel<<<(int)(M / 256), 256, 0, stream>>>(x, ln_g, ln_b, xs);

  // in-proj: (M x 96) x (96 x 384) -> xz bf16 ; 16x64 strip per wave
  {
    int tot = (int)(M / 16) * (384 / 64);
    gemm_bf16<4, 3><<<tot / 8, 256, 0, stream>>>(xs, in_wt, nullptr, xzb,
                                                 (int)M, 384, 96, nullptr, LL, CC);
  }
  conv_silu<<<(int)(M * DIN / 256), 256, 0, stream>>>(xzb, conv_w, conv_b, xcb);

  // x-proj: (M x 192) x (192 x 48) -> x_dbl fp32 ; whole N strip per wave
  {
    int tot = (int)(M / 16) * (48 / 48);
    gemm_bf16<3, 0><<<tot / 8, 256, 0, stream>>>(xcb, xprojt, xdbl, nullptr,
                                                 (int)M, 48, 192, nullptr, LL, CC);
  }
  dt_prep<<<(int)(M / 256), 256, 0, stream>>>(xdbl, dtin);

  // dt-proj: (M x 32) x (32 x 192) -> softplus(. + b) fp32 ; 16x64 per wave
  {
    int tot = (int)(M / 16) * (192 / 64);
    gemm_bf16<4, 1><<<tot / 8, 256, 0, stream>>>(dtin, dtprjt, dtb, nullptr,
                                                 (int)M, 192, 32, dtproj_b, LL, CC);
  }

  scan_pass1<<<NB * NCHUNK, DIN, 0, stream>>>(dtb, xcb, xdbl, Aneg, Pc, Sc);
  scan_pass2<<<NB * DIN * DST / 256, 256, 0, stream>>>(Pc, Sc, Hinit);
  scan_pass3<<<NB * NCHUNK, DIN, 0, stream>>>(dtb, xcb, xdbl, Aneg, Hinit, xzb,
                                              D_skip, ybf);

  // out-proj: (M x 192) x (192 x 96) -> transposed store (n, C, L) ; 16x48/wave
  {
    int tot = (int)(M / 16) * (96 / 48);
    gemm_bf16<3, 2><<<tot / 8, 256, 0, stream>>>(ybf, out_wt, out, nullptr,
                                                 (int)M, 96, 192, nullptr, LL, CC);
  }
}
